// DCTNet_16183436771390
// MI455X (gfx1250) — compile-verified, matrix operand order checked by reading, not run on hardware
//
#include <hip/hip_runtime.h>
#include <hip/hip_bf16.h>

typedef __attribute__((ext_vector_type(16))) _Float16 v16h;
typedef __attribute__((ext_vector_type(8)))  _Float16 v8h;
typedef __attribute__((ext_vector_type(8)))  float    v8f;

#define BN_EPS 1e-3f
#define GAMMA_C 1000000.0f

__device__ __forceinline__ float sigf(float t) { return 1.0f / (1.0f + __expf(-t)); }

// ---------------- DCT: [256,256,256,1] -> [256,1024,64] f32 ----------------
__global__ void dct_kernel(const float* __restrict__ inp, const float* __restrict__ basis,
                           float* __restrict__ dct) {
    __shared__ float bs[64 * 64];
    for (int i = threadIdx.x; i < 64 * 64; i += blockDim.x) bs[i] = basis[i];
    __syncthreads();
    int idx = blockIdx.x * blockDim.x + threadIdx.x;   // patch id, 256*1024 total
    int b = idx >> 10;
    int r = idx & 1023;
    int by = r >> 5, bx = r & 31;
    float patch[64];
    const float* src = inp + ((size_t)b * 256 + by * 8) * 256 + bx * 8;
#pragma unroll
    for (int dy = 0; dy < 8; ++dy)
#pragma unroll
        for (int dx = 0; dx < 8; ++dx)
            patch[dy * 8 + dx] = src[(size_t)dy * 256 + dx];
    float* out = dct + (size_t)idx * 64;
    for (int n = 0; n < 64; ++n) {
        float acc = 0.f;
#pragma unroll
        for (int k = 0; k < 64; ++k) acc += patch[k] * bs[k * 64 + n];
        out[n] = acc;
    }
}

// ------- Soft histogram (exact, deterministic): dct -> x0 [256,120,64] f32 -------
// Per value v, only bins k-1,k,k+1 (k=floor(v+60)) have non-saturated sigmoid diffs.
__global__ void hist_kernel(const float* __restrict__ dct, float* __restrict__ x0) {
    __shared__ float bins[64 * 121];
    float* mybins = bins + threadIdx.x * 121;
    for (int i = 0; i < 121; ++i) mybins[i] = 0.f;
    int gid = blockIdx.x * 64 + threadIdx.x;  // (b,c)
    int b = gid >> 6, c = gid & 63;
    const float* src = dct + (size_t)b * 1024 * 64 + c;
    for (int p = 0; p < 1024; ++p) {
        float v = src[(size_t)p * 64];
        int k = (int)floorf(v + 60.0f);
#pragma unroll
        for (int d = -1; d <= 1; ++d) {
            int j = k + d;
            if (j >= 0 && j <= 119) {
                float lo = (float)(j - 60);
                mybins[j] += sigf(GAMMA_C * (v - lo)) - sigf(GAMMA_C * (v - (lo + 1.0f)));
            }
        }
    }
    for (int i = 0; i < 120; ++i)
        x0[((size_t)b * 120 + i) * 64 + c] = mybins[i] * (1.0f / 1024.0f);
}

// ------------- weight transpose+f16: w[tap][cin][cout] -> wt[tap][cout][cin] -------------
__global__ void prep_weights(const float* __restrict__ w, _Float16* __restrict__ wt,
                             int taps, int cin, int cout) {
    int idx = blockIdx.x * blockDim.x + threadIdx.x;
    int total = taps * cin * cout;
    if (idx >= total) return;
    int tap = idx / (cin * cout);
    int r = idx % (cin * cout);
    int n = r / cin, c = r % cin;
    wt[idx] = (_Float16)w[((size_t)tap * cin + c) * cout + n];
}

// per-channel fused epilogue: y = acc*scale + shift; scale=g*rsqrt(v+eps), shift=(bias-m)*scale+beta
__global__ void prep_params(const float* __restrict__ bias, const float* __restrict__ g,
                            const float* __restrict__ beta, const float* __restrict__ m,
                            const float* __restrict__ v, float* __restrict__ scale,
                            float* __restrict__ shift, int n) {
    int i = blockIdx.x * blockDim.x + threadIdx.x;
    if (i >= n) return;
    float s = g[i] * rsqrtf(v[i] + BN_EPS);
    scale[i] = s;
    shift[i] = (bias[i] - m[i]) * s + beta[i];
}

// ---------------- block0 conv1: 3x3, cin=1 -> 64, f32 VALU, fused BN+ReLU ----------------
__global__ void conv3x3_c1(const float* __restrict__ x, const float* __restrict__ w,
                           const float* __restrict__ scale, const float* __restrict__ shift,
                           _Float16* __restrict__ out, int B, int H, int W) {
    __shared__ float wl[9 * 64];
    __shared__ float sc[64], sh[64];
    for (int i = threadIdx.x; i < 9 * 64; i += blockDim.x) wl[i] = w[i];
    for (int i = threadIdx.x; i < 64; i += blockDim.x) { sc[i] = scale[i]; sh[i] = shift[i]; }
    __syncthreads();
    int idx = blockIdx.x * blockDim.x + threadIdx.x;
    int P = B * H * W;
    if (idx >= P) return;
    int w_ = idx % W; int t = idx / W; int h_ = t % H; int b_ = t / H;
    float vin[9];
#pragma unroll
    for (int tap = 0; tap < 9; ++tap) {
        int dy = tap / 3 - 1, dx = tap % 3 - 1;
        int hs = h_ + dy, ws2 = w_ + dx;
        vin[tap] = (hs >= 0 && hs < H && ws2 >= 0 && ws2 < W)
                   ? x[((size_t)b_ * H + hs) * W + ws2] : 0.f;
    }
    _Float16* o = out + (size_t)idx * 64;
    for (int n = 0; n < 64; ++n) {
        float acc = 0.f;
#pragma unroll
        for (int tap = 0; tap < 9; ++tap) acc += vin[tap] * wl[tap * 64 + n];
        float y = acc * sc[n] + sh[n];
        o[n] = (_Float16)fmaxf(y, 0.f);
    }
}

// -------- WMMA implicit-GEMM conv (taps=1 or 9), f16 in, f32 acc, fused BN+ReLU --------
// 2x2 register-blocked: one wave computes a 32(M)x32(N) output tile (4 C fragments).
// Per K-step: load 2 A frags + 2 B frags, issue 4 WMMAs (2x arithmetic intensity vs 1x1,
// 4 independent accumulator chains hide WMMA latency).
// act: [B*H*W, cin] f16 row-major; wt: [taps][cout][cin] f16; out: [B*H*W, cout] f16
// A/B fragment layout per CDNA5 ISA 16-bit 16x32:
// lane l holds row/col l&15; K elems j<8 -> K=k0+(l>=16?8:0)+j, j>=8 -> +16.
__global__ void wmma_conv(const _Float16* __restrict__ act, const _Float16* __restrict__ wt,
                          const float* __restrict__ scale, const float* __restrict__ shift,
                          _Float16* __restrict__ out,
                          int H, int W, int cin, int cout, int taps) {
    int lane = threadIdx.x;
    int row = lane & 15;
    int hi = lane >> 4;
    int tileM = blockIdx.x * 32;
    int n0 = blockIdx.y * 32;
    int ka0 = hi * 8;

    // decode the two A rows handled by this lane
    int m0 = tileM + row;
    int m1 = tileM + 16 + row;
    int w0 = m0 % W; int t0 = m0 / W; int h0 = t0 % H; int b0 = t0 / H;
    int w1 = m1 % W; int t1 = m1 / W; int h1 = t1 % H; int b1 = t1 / H;
    // the two B columns handled by this lane
    int nB0 = n0 + row;
    int nB1 = n0 + 16 + row;

    v8f acc00 = {}, acc01 = {}, acc10 = {}, acc11 = {};

    for (int tap = 0; tap < taps; ++tap) {
        int dy = (taps == 9) ? (tap / 3 - 1) : 0;
        int dx = (taps == 9) ? (tap % 3 - 1) : 0;
        int hs0 = h0 + dy, ws0 = w0 + dx;
        int hs1 = h1 + dy, ws1 = w1 + dx;
        bool valid0 = (hs0 >= 0) && (hs0 < H) && (ws0 >= 0) && (ws0 < W);
        bool valid1 = (hs1 >= 0) && (hs1 < H) && (ws1 >= 0) && (ws1 < W);
        const _Float16* arow0 = act + (((size_t)b0 * H + hs0) * W + ws0) * cin;
        const _Float16* arow1 = act + (((size_t)b1 * H + hs1) * W + ws1) * cin;
        const _Float16* brow0 = wt + ((size_t)tap * cout + nB0) * cin;
        const _Float16* brow1 = wt + ((size_t)tap * cout + nB1) * cin;
        for (int k0 = 0; k0 < cin; k0 += 32) {
            v16h a0 = {}, a1 = {};
            if (valid0) {
                v8h lo = *(const v8h*)(arow0 + k0 + ka0);
                v8h hi8 = *(const v8h*)(arow0 + k0 + ka0 + 16);
#pragma unroll
                for (int j = 0; j < 8; ++j) { a0[j] = lo[j]; a0[j + 8] = hi8[j]; }
            }
            if (valid1) {
                v8h lo = *(const v8h*)(arow1 + k0 + ka0);
                v8h hi8 = *(const v8h*)(arow1 + k0 + ka0 + 16);
#pragma unroll
                for (int j = 0; j < 8; ++j) { a1[j] = lo[j]; a1[j + 8] = hi8[j]; }
            }
            v16h bf0, bf1;
            {
                v8h lo = *(const v8h*)(brow0 + k0 + ka0);
                v8h hi8 = *(const v8h*)(brow0 + k0 + ka0 + 16);
#pragma unroll
                for (int j = 0; j < 8; ++j) { bf0[j] = lo[j]; bf0[j + 8] = hi8[j]; }
            }
            {
                v8h lo = *(const v8h*)(brow1 + k0 + ka0);
                v8h hi8 = *(const v8h*)(brow1 + k0 + ka0 + 16);
#pragma unroll
                for (int j = 0; j < 8; ++j) { bf1[j] = lo[j]; bf1[j + 8] = hi8[j]; }
            }
            acc00 = __builtin_amdgcn_wmma_f32_16x16x32_f16(false, a0, false, bf0,
                                                           (short)0, acc00, false, false);
            acc01 = __builtin_amdgcn_wmma_f32_16x16x32_f16(false, a0, false, bf1,
                                                           (short)0, acc01, false, false);
            acc10 = __builtin_amdgcn_wmma_f32_16x16x32_f16(false, a1, false, bf0,
                                                           (short)0, acc10, false, false);
            acc11 = __builtin_amdgcn_wmma_f32_16x16x32_f16(false, a1, false, bf1,
                                                           (short)0, acc11, false, false);
        }
    }
    // epilogue: C/D layout -> VGPR r holds row r + (lane>=16 ? 8 : 0), col = lane&15
    float sc0 = scale[nB0], sf0 = shift[nB0];
    float sc1 = scale[nB1], sf1 = shift[nB1];
#pragma unroll
    for (int r = 0; r < 8; ++r) {
        int mmA = tileM + hi * 8 + r;       // rows of acc00/acc01
        int mmB = mmA + 16;                 // rows of acc10/acc11
        out[(size_t)mmA * cout + nB0] = (_Float16)fmaxf(acc00[r] * sc0 + sf0, 0.f);
        out[(size_t)mmA * cout + nB1] = (_Float16)fmaxf(acc01[r] * sc1 + sf1, 0.f);
        out[(size_t)mmB * cout + nB0] = (_Float16)fmaxf(acc10[r] * sc0 + sf0, 0.f);
        out[(size_t)mmB * cout + nB1] = (_Float16)fmaxf(acc11[r] * sc1 + sf1, 0.f);
    }
}

// ---------------- 2x2 maxpool, f16 ----------------
__global__ void maxpool_kernel(const _Float16* __restrict__ in, _Float16* __restrict__ out,
                               int B, int H, int W, int C) {
    int Ho = H >> 1, Wo = W >> 1;
    size_t total = (size_t)B * Ho * Wo * C;
    size_t idx = (size_t)blockIdx.x * blockDim.x + threadIdx.x;
    if (idx >= total) return;
    int c = (int)(idx % C); size_t t = idx / C;
    int wo = (int)(t % Wo); t /= Wo;
    int ho = (int)(t % Ho); int b = (int)(t / Ho);
    const _Float16* p = in + (((size_t)b * H + ho * 2) * W + wo * 2) * C + c;
    float a0 = (float)p[0], a1 = (float)p[C];
    float a2 = (float)p[(size_t)W * C], a3 = (float)p[(size_t)W * C + C];
    out[idx] = (_Float16)fmaxf(fmaxf(a0, a1), fmaxf(a2, a3));
}

// ---------------- GAP + FC + softmax -> d_out [256,4] f32 ----------------
__global__ void gap_fc_softmax(const _Float16* __restrict__ act, const float* __restrict__ fc,
                               float* __restrict__ out, int P, int C) {
    __shared__ float g[512];
    __shared__ float logit[4];
    int b = blockIdx.x;
    int c = threadIdx.x;
    if (c < C) {
        float s = 0.f;
        for (int p = 0; p < P; ++p) s += (float)act[((size_t)b * P + p) * C + c];
        g[c] = s / (float)P;
    }
    __syncthreads();
    if (c < 4) {
        float s = 0.f;
        for (int k = 0; k < C; ++k) s += g[k] * fc[k * 4 + c];
        logit[c] = s;
    }
    __syncthreads();
    if (c == 0) {
        float mx = fmaxf(fmaxf(logit[0], logit[1]), fmaxf(logit[2], logit[3]));
        float e0 = __expf(logit[0] - mx), e1 = __expf(logit[1] - mx);
        float e2 = __expf(logit[2] - mx), e3 = __expf(logit[3] - mx);
        float inv = 1.f / (e0 + e1 + e2 + e3);
        out[b * 4 + 0] = e0 * inv; out[b * 4 + 1] = e1 * inv;
        out[b * 4 + 2] = e2 * inv; out[b * 4 + 3] = e3 * inv;
    }
}

extern "C" void kernel_launch(void* const* d_in, const int* in_sizes, int n_in,
                              void* d_out, int out_size, void* d_ws, size_t ws_size,
                              hipStream_t stream) {
    (void)out_size; (void)ws_size;
    // Detect flattening order of the inputs dict.
    const float *inp, *basis;
    if (in_sizes[0] == 4096) { basis = (const float*)d_in[0]; inp = (const float*)d_in[1]; }
    else                     { inp = (const float*)d_in[0]; basis = (const float*)d_in[1]; }
    const float* fc = (const float*)d_in[n_in - 1];

    struct BP { const float *w1,*b1,*g1,*be1,*m1,*v1,*w2,*b2,*g2,*be2,*m2,*v2; } bp[4];
    int fs[4] = {64, 128, 256, 512}, cs[4] = {1, 64, 128, 256};
    int p = 2;
    for (int i = 0; i < 4; ++i) {
        void* const* a = d_in + p;
        int w1n = 9 * cs[i] * fs[i];
        if (in_sizes[p] == w1n) {  // insertion order: w1,b1,g1,b,m,v,w2,b2,g2,b,m,v
            bp[i].w1 = (const float*)a[0];  bp[i].b1 = (const float*)a[1];
            bp[i].g1 = (const float*)a[2];  bp[i].be1 = (const float*)a[3];
            bp[i].m1 = (const float*)a[4];  bp[i].v1 = (const float*)a[5];
            bp[i].w2 = (const float*)a[6];  bp[i].b2 = (const float*)a[7];
            bp[i].g2 = (const float*)a[8];  bp[i].be2 = (const float*)a[9];
            bp[i].m2 = (const float*)a[10]; bp[i].v2 = (const float*)a[11];
        } else {                   // sorted: b1,b2,bn1_b,bn1_g,bn1_m,bn1_v,bn2_*,w1,w2
            bp[i].b1 = (const float*)a[0];  bp[i].b2 = (const float*)a[1];
            bp[i].be1 = (const float*)a[2]; bp[i].g1 = (const float*)a[3];
            bp[i].m1 = (const float*)a[4];  bp[i].v1 = (const float*)a[5];
            bp[i].be2 = (const float*)a[6]; bp[i].g2 = (const float*)a[7];
            bp[i].m2 = (const float*)a[8];  bp[i].v2 = (const float*)a[9];
            bp[i].w1 = (const float*)a[10]; bp[i].w2 = (const float*)a[11];
        }
        p += 12;
    }

    const size_t MB = 1024ull * 1024ull;
    char* ws = (char*)d_ws;
    float* x0      = (float*)(ws);                       // 7.9 MB
    _Float16* pA   = (_Float16*)(ws + 8 * MB);           // 252 MB region
    _Float16* pB   = (_Float16*)(ws + (8 + 252) * MB);   // 252 MB region
    float* dct     = (float*)pA;                         // 67 MB, dead before pA is used
    _Float16* wt1  = (_Float16*)(ws + (8 + 504) * MB);   // 4 MB
    _Float16* wt2  = (_Float16*)(ws + (8 + 508) * MB);   // 1 MB
    float* sc1     = (float*)(ws + (8 + 509) * MB);
    float* sh1 = sc1 + 512; float* sc2 = sh1 + 512; float* sh2 = sc2 + 512;

    const int B = 256;
    dct_kernel<<<(256 * 1024) / 256, 256, 0, stream>>>(inp, basis, dct);
    hist_kernel<<<(B * 64) / 64, 64, 0, stream>>>(dct, x0);

    int H = 120, W = 64;
    // block0: 3x3 (cin=1) direct f32, then 1x1 64->64 via WMMA
    prep_params<<<1, 64, 0, stream>>>(bp[0].b1, bp[0].g1, bp[0].be1, bp[0].m1, bp[0].v1, sc1, sh1, 64);
    conv3x3_c1<<<(B * H * W + 255) / 256, 256, 0, stream>>>(x0, bp[0].w1, sc1, sh1, pA, B, H, W);
    prep_weights<<<(64 * 64 + 255) / 256, 256, 0, stream>>>(bp[0].w2, wt2, 1, 64, 64);
    prep_params<<<1, 64, 0, stream>>>(bp[0].b2, bp[0].g2, bp[0].be2, bp[0].m2, bp[0].v2, sc2, sh2, 64);
    { dim3 g(B * H * W / 32, 64 / 32); wmma_conv<<<g, 32, 0, stream>>>(pA, wt2, sc2, sh2, pB, H, W, 64, 64, 1); }
    maxpool_kernel<<<(unsigned)(((size_t)B * (H/2) * (W/2) * 64 + 255) / 256), 256, 0, stream>>>(pB, pA, B, H, W, 64);
    H /= 2; W /= 2;

    _Float16* cur = pA; _Float16* oth = pB;
    for (int i = 1; i < 4; ++i) {
        int cin = cs[i], f = fs[i];
        prep_weights<<<(9 * cin * f + 255) / 256, 256, 0, stream>>>(bp[i].w1, wt1, 9, cin, f);
        prep_params<<<(f + 255) / 256, 256, 0, stream>>>(bp[i].b1, bp[i].g1, bp[i].be1, bp[i].m1, bp[i].v1, sc1, sh1, f);
        { dim3 g(B * H * W / 32, f / 32); wmma_conv<<<g, 32, 0, stream>>>(cur, wt1, sc1, sh1, oth, H, W, cin, f, 9); }
        prep_weights<<<(f * f + 255) / 256, 256, 0, stream>>>(bp[i].w2, wt2, 1, f, f);
        prep_params<<<(f + 255) / 256, 256, 0, stream>>>(bp[i].b2, bp[i].g2, bp[i].be2, bp[i].m2, bp[i].v2, sc2, sh2, f);
        { dim3 g(B * H * W / 32, f / 32); wmma_conv<<<g, 32, 0, stream>>>(oth, wt2, sc2, sh2, cur, H, W, f, f, 1); }
        maxpool_kernel<<<(unsigned)(((size_t)B * (H/2) * (W/2) * f + 255) / 256), 256, 0, stream>>>(cur, oth, B, H, W, f);
        H /= 2; W /= 2;
        _Float16* tmp = cur; cur = oth; oth = tmp;
    }
    // cur now holds [256, 7, 4, 512] f16
    gap_fc_softmax<<<B, 512, 0, stream>>>(cur, fc, (float*)d_out, H * W, 512);
}
